// SimplifiedLlama4Attention_78932908965940
// MI455X (gfx1250) — matrix operand order repaired and checked by reading, not compiled, not run on hardware
//
#include <hip/hip_runtime.h>
#include <hip/hip_bf16.h>
#include <stdint.h>

// ---------------------------------------------------------------------------
// SimplifiedLlama4Attention for MI455X (gfx1250), bf16 WMMA pipeline.
// B=2, S=2048, H=2048, NH=16, NKV=4, HD=128.
// d_out = out[B,S,H] fp32  ++  attn[B,NH,S,S] fp32.
// Workspace requirement ~432 MB.
// ---------------------------------------------------------------------------

typedef __bf16 bf16;
typedef __attribute__((ext_vector_type(16))) __bf16 v16bf;
typedef __attribute__((ext_vector_type(8)))  float  v8f;

#define BB   2
#define SS   2048
#define HH   2048
#define NHQ  16
#define NKV  4
#define HD   128
#define HALF 64
#define MTOT (BB * SS)        // 4096 rows of hidden

// A-matrix (16x32 bf16) element->K mapping used here: contiguous k = 16*hi + e (B uses the same).
__device__ __forceinline__ int kmapA(int e, int hi) {
    return hi * 16 + e;
}

// ---------------------------------------------------------------------------
// fp32 -> bf16 elementwise
// ---------------------------------------------------------------------------
__global__ void cvt_f32_bf16(const float* __restrict__ src, bf16* __restrict__ dst, size_t n) {
    size_t i = (size_t)blockIdx.x * blockDim.x + threadIdx.x;
    if (i < n) dst[i] = (bf16)src[i];
}

// ---------------------------------------------------------------------------
// Generic bf16 GEMM: C[M,N] = A[M,K] * B[K,N], row-major, fp32 accumulate.
// 64x64 C-tile per block, 128 threads (4 waves), BK=32, WMMA 16x16x32 bf16.
// A tile is brought in with GLOBAL_LOAD_ASYNC_TO_LDS_B128 (ASYNCcnt).
// ---------------------------------------------------------------------------
#define BM 64
#define BN 64
#define BK 32

__global__ __launch_bounds__(128)
void gemm_bf16_wmma(const bf16* __restrict__ A, const bf16* __restrict__ Bm,
                    float* __restrict__ C,
                    int M, int N, int K, int lda, int ldb, int ldc)
{
    __shared__ __align__(16) bf16 As[BM][BK + 8];   // [row][k], stride 40 (80B)
    __shared__ __align__(16) bf16 Bs[BN][BK + 8];   // transposed: [n][k]

    const int tid  = threadIdx.x;
    const int wave = tid >> 5;
    const int lane = tid & 31;
    const int ln   = lane & 15;
    const int hi   = lane >> 4;
    const int rowBase = blockIdx.y * BM;
    const int colBase = blockIdx.x * BN;
    const int r0 = wave * 16;               // this wave's 16-row stripe

    // precompute this thread's two A-tile chunk coordinates (16B each)
    const int c0   = tid * 2;
    const int row0 = c0 >> 2,        col0 = (c0 & 3) * 8;
    const int row1 = (c0 + 1) >> 2,  col1 = ((c0 + 1) & 3) * 8;
    const uint32_t ldsA0 = (uint32_t)(uintptr_t)&As[row0][col0];
    const uint32_t ldsA1 = (uint32_t)(uintptr_t)&As[row1][col1];

    v8f acc[4];
    for (int j = 0; j < 4; ++j)
        for (int e = 0; e < 8; ++e) acc[j][e] = 0.0f;

    for (int kk = 0; kk < K; kk += BK) {
        // ---- A tile 64x32: async global->LDS, 16B per lane per op ----
        {
            uint64_t g0 = (uint64_t)(uintptr_t)(A + (size_t)(rowBase + row0) * lda + kk + col0);
            uint64_t g1 = (uint64_t)(uintptr_t)(A + (size_t)(rowBase + row1) * lda + kk + col1);
            asm volatile("global_load_async_to_lds_b128 %0, %1, off"
                         :: "v"(ldsA0), "v"(g0) : "memory");
            asm volatile("global_load_async_to_lds_b128 %0, %1, off"
                         :: "v"(ldsA1), "v"(g1) : "memory");
        }
        // ---- B tile 32x64, store transposed into Bs[n][k] ----
        for (int i = 0; i < 2; ++i) {
            int c = tid * 2 + i;                  // 0..255
            int krow = c >> 3, n8 = (c & 7) * 8;
            const bf16* gp = Bm + (size_t)(kk + krow) * ldb + colBase + n8;
            #pragma unroll
            for (int t = 0; t < 8; ++t) Bs[n8 + t][krow] = gp[t];
        }
        if (kk + BK < K)
            __builtin_prefetch(Bm + (size_t)(kk + BK + (tid >> 3)) * ldb + colBase, 0, 0);

        asm volatile("s_wait_asynccnt 0x0" ::: "memory");
        __syncthreads();

        // ---- A fragment for this wave's 16 rows ----
        v16bf af;
        #pragma unroll
        for (int e = 0; e < 16; ++e) af[e] = As[r0 + ln][kmapA(e, hi)];

        // ---- 4 column subtiles ----
        #pragma unroll
        for (int j = 0; j < 4; ++j) {
            v16bf bf;
            #pragma unroll
            for (int e = 0; e < 16; ++e) bf[e] = Bs[j * 16 + ln][hi * 16 + e];
            acc[j] = __builtin_amdgcn_wmma_f32_16x16x32_bf16(
                false, af, false, bf, (short)0, acc[j], false, false);
        }
        __syncthreads();
    }

    // ---- store C: lane holds N=ln, M=hi*8+v ----
    #pragma unroll
    for (int j = 0; j < 4; ++j)
        #pragma unroll
        for (int v = 0; v < 8; ++v)
            C[(size_t)(rowBase + r0 + hi * 8 + v) * ldc + colBase + j * 16 + ln] = acc[j][v];
}

// ---------------------------------------------------------------------------
// RoPE (interleaved pairs) + l2norm over head dim, rearrange to head-major bf16.
// ---------------------------------------------------------------------------
__global__ __launch_bounds__(64)
void rope_l2norm(const float* __restrict__ src, bf16* __restrict__ dst,
                 const float* __restrict__ cosT, const float* __restrict__ sinT,
                 int n_heads)
{
    int idx  = blockIdx.x;
    int head = idx % n_heads;
    int s    = (idx / n_heads) % SS;
    int b    = idx / (n_heads * SS);
    int t    = threadIdx.x;

    const float* x = src + ((size_t)(b * SS + s) * n_heads + head) * HD;
    float x0 = x[2 * t], x1 = x[2 * t + 1];
    float c  = cosT[(size_t)s * HALF + t];
    float sn = sinT[(size_t)s * HALF + t];
    float o0 = x0 * c - x1 * sn;
    float o1 = x0 * sn + x1 * c;

    __shared__ float red[64];
    red[t] = o0 * o0 + o1 * o1;
    __syncthreads();
    for (int st = 32; st > 0; st >>= 1) {
        if (t < st) red[t] += red[t + st];
        __syncthreads();
    }
    float scale = rsqrtf(red[0] * (1.0f / HD) + 1e-6f);

    bf16* d = dst + ((size_t)(b * n_heads + head) * SS + s) * HD;
    d[2 * t]     = (bf16)(o0 * scale);
    d[2 * t + 1] = (bf16)(o1 * scale);
}

// ---------------------------------------------------------------------------
// V: fp32 [B,S,NKV*HD] -> bf16 head-major [B,NKV,S,HD]
// ---------------------------------------------------------------------------
__global__ void rearrange_v(const float* __restrict__ Vf, bf16* __restrict__ Vbh) {
    size_t idx = (size_t)blockIdx.x * blockDim.x + threadIdx.x;
    size_t total = (size_t)BB * SS * NKV * HD;
    if (idx >= total) return;
    int d  = idx % HD;
    int kv = (idx / HD) % NKV;
    size_t bs = idx / (HD * NKV);
    int s = bs % SS, b = bs / SS;
    Vbh[((size_t)(b * NKV + kv) * SS + s) * HD + d] = (bf16)Vf[idx];
}

// ---------------------------------------------------------------------------
// Fused scores + softmax per (b, h, 16-row Q stripe).
// 4 waves split the 128 column blocks; WMMA scores land in a 16x2048 fp32 LDS
// stripe (128 KB, fits the 320 KB WGP LDS); cooperative softmax in LDS; single
// write of fp32 probs (d_out) and bf16 probs (for the PV GEMM).
// Saves a full 537 MB read + 537 MB write vs. separate scores/softmax passes.
// ---------------------------------------------------------------------------
__global__ __launch_bounds__(128)
void attn_scores_softmax(const bf16* __restrict__ Q, const bf16* __restrict__ Kh,
                         float* __restrict__ attn, bf16* __restrict__ Pb)
{
    __shared__ float Srow[16][SS];          // 128 KB score stripe
    __shared__ float redm[16][8];
    __shared__ float reds[16][8];
    __shared__ float invs[16];

    const int qb = blockIdx.x, h = blockIdx.y, b = blockIdx.z;
    const int tid = threadIdx.x, wave = tid >> 5, lane = tid & 31;
    const int ln = lane & 15, hi = lane >> 4;
    const int kv = h / (NHQ / NKV);

    const bf16* qbase = Q  + ((size_t)(b * NHQ + h) * SS + qb * 16) * HD;
    const bf16* kbase = Kh + (size_t)(b * NKV + kv) * SS * HD;

    // A fragments over HD=128 (4 k-steps), identical in all 4 waves
    v16bf aq[4];
    #pragma unroll
    for (int t = 0; t < 4; ++t)
        #pragma unroll
        for (int e = 0; e < 16; ++e)
            aq[t][e] = qbase[(size_t)ln * HD + t * 32 + kmapA(e, hi)];

    const float scale = 0.08838834764831845f;   // 1/sqrt(128)

    // each wave handles column blocks cb = wave, wave+4, ...
    for (int cb = wave; cb < SS / 16; cb += 4) {
        v8f acc;
        for (int e = 0; e < 8; ++e) acc[e] = 0.0f;
        #pragma unroll
        for (int t = 0; t < 4; ++t) {
            v16bf bk = *(const v16bf*)(kbase + (size_t)(cb * 16 + ln) * HD + t * 32 + hi * 16);
            acc = __builtin_amdgcn_wmma_f32_16x16x32_bf16(
                false, aq[t], false, bk, (short)0, acc, false, false);
        }
        const int kcol = cb * 16 + ln;
        #pragma unroll
        for (int v = 0; v < 8; ++v) {
            int qrow = qb * 16 + hi * 8 + v;
            Srow[hi * 8 + v][kcol] = acc[v] * scale + (kcol > qrow ? -1e9f : 0.0f);
        }
    }
    __syncthreads();

    // ---- softmax: 8 threads per row, 256 columns per thread ----
    const int row = tid >> 3, seg = tid & 7;
    float* r = &Srow[row][seg * 256];

    float m = -3.4e38f;
    for (int i = 0; i < 256; ++i) m = fmaxf(m, r[i]);
    redm[row][seg] = m;
    __syncthreads();
    if (seg == 0) {
        float mm = redm[row][0];
        #pragma unroll
        for (int j = 1; j < 8; ++j) mm = fmaxf(mm, redm[row][j]);
        redm[row][0] = mm;
    }
    __syncthreads();
    m = redm[row][0];

    float sum = 0.0f;
    for (int i = 0; i < 256; ++i) { float e = __expf(r[i] - m); r[i] = e; sum += e; }
    reds[row][seg] = sum;
    __syncthreads();
    if (seg == 0) {
        float ss = reds[row][0];
        #pragma unroll
        for (int j = 1; j < 8; ++j) ss += reds[row][j];
        invs[row] = 1.0f / ss;
    }
    __syncthreads();

    // ---- coalesced writeback: fp32 probs to d_out, bf16 probs for PV ----
    for (int r2 = 0; r2 < 16; ++r2) {
        const float inv = invs[r2];
        const size_t rowIdx = (size_t)(b * NHQ + h) * SS + qb * 16 + r2;
        float* dst  = attn + rowIdx * SS;
        bf16*  dstb = Pb   + rowIdx * SS;
        for (int c = tid; c < SS; c += 128) {
            float pr = Srow[r2][c] * inv;
            dst[c]  = pr;
            dstb[c] = (bf16)pr;
        }
    }
}

// ---------------------------------------------------------------------------
// Host-side orchestration
// ---------------------------------------------------------------------------
extern "C" void kernel_launch(void* const* d_in, const int* in_sizes, int n_in,
                              void* d_out, int out_size, void* d_ws, size_t ws_size,
                              hipStream_t stream) {
    (void)in_sizes; (void)n_in; (void)out_size; (void)ws_size;

    const float* hid = (const float*)d_in[0];
    // d_in[1] attention_mask, d_in[2] position_ids: applied analytically
    const float* Wq  = (const float*)d_in[3];
    const float* Wk  = (const float*)d_in[4];
    const float* Wv  = (const float*)d_in[5];
    const float* Wo  = (const float*)d_in[6];
    const float* rc  = (const float*)d_in[7];
    const float* rs  = (const float*)d_in[8];

    float* out  = (float*)d_out;                        // [B,S,H]
    float* attn = out + (size_t)BB * SS * HH;           // [B,NH,S,S]

    // ---- workspace carve-up (256B aligned) ----
    char* w = (char*)d_ws;
    auto carve = [&](size_t bytes) { void* p = (void*)w; w += (bytes + 255) & ~(size_t)255; return p; };
    const size_t nX = (size_t)MTOT * HH;
    bf16* Xb  = (bf16*)carve(nX * 2);
    bf16* Wqb = (bf16*)carve((size_t)HH * HH * 2);
    bf16* Wkb = (bf16*)carve((size_t)HH * (NKV * HD) * 2);
    bf16* Wvb = (bf16*)carve((size_t)HH * (NKV * HD) * 2);
    bf16* Wob = (bf16*)carve((size_t)HH * HH * 2);
    float* Qf = (float*)carve((size_t)MTOT * (NHQ * HD) * 4);
    float* Kf = (float*)carve((size_t)MTOT * (NKV * HD) * 4);
    float* Vf = (float*)carve((size_t)MTOT * (NKV * HD) * 4);
    bf16* Qbh = (bf16*)carve((size_t)BB * NHQ * SS * HD * 2);
    bf16* Kbh = (bf16*)carve((size_t)BB * NKV * SS * HD * 2);
    bf16* Vbh = (bf16*)carve((size_t)BB * NKV * SS * HD * 2);
    bf16* Pb  = (bf16*)carve((size_t)BB * NHQ * SS * SS * 2);   // 268 MB
    float* Ob = (float*)carve((size_t)MTOT * HH * 4);
    bf16* Obb = (bf16*)carve((size_t)MTOT * HH * 2);

    // ---- 1. fp32 -> bf16 conversions ----
    auto cvt = [&](const float* s, bf16* d, size_t n) {
        cvt_f32_bf16<<<(unsigned)((n + 255) / 256), 256, 0, stream>>>(s, d, n);
    };
    cvt(hid, Xb, nX);
    cvt(Wq, Wqb, (size_t)HH * HH);
    cvt(Wk, Wkb, (size_t)HH * NKV * HD);
    cvt(Wv, Wvb, (size_t)HH * NKV * HD);
    cvt(Wo, Wob, (size_t)HH * HH);

    // ---- 2. Q/K/V projections ----
    gemm_bf16_wmma<<<dim3(HH / BN, MTOT / BM), 128, 0, stream>>>(
        Xb, Wqb, Qf, MTOT, HH, HH, HH, HH, HH);
    gemm_bf16_wmma<<<dim3((NKV * HD) / BN, MTOT / BM), 128, 0, stream>>>(
        Xb, Wkb, Kf, MTOT, NKV * HD, HH, HH, NKV * HD, NKV * HD);
    gemm_bf16_wmma<<<dim3((NKV * HD) / BN, MTOT / BM), 128, 0, stream>>>(
        Xb, Wvb, Vf, MTOT, NKV * HD, HH, HH, NKV * HD, NKV * HD);

    // ---- 3. RoPE + l2norm, rearrange ----
    rope_l2norm<<<BB * SS * NHQ, 64, 0, stream>>>(Qf, Qbh, rc, rs, NHQ);
    rope_l2norm<<<BB * SS * NKV, 64, 0, stream>>>(Kf, Kbh, rc, rs, NKV);
    rearrange_v<<<(unsigned)(((size_t)BB * SS * NKV * HD + 255) / 256), 256, 0, stream>>>(Vf, Vbh);

    // ---- 4. fused scores + mask + softmax ----
    attn_scores_softmax<<<dim3(SS / 16, NHQ, BB), 128, 0, stream>>>(Qbh, Kbh, attn, Pb);

    // ---- 5. PV: out_h = P[b,h] @ V[b,kv]  (one GEMM launch per head) ----
    for (int z = 0; z < BB * NHQ; ++z) {
        int b = z / NHQ, h = z % NHQ, kv = h / (NHQ / NKV);
        gemm_bf16_wmma<<<dim3(HD / BN, SS / BM), 128, 0, stream>>>(
            Pb + (size_t)z * SS * SS,
            Vbh + (size_t)(b * NKV + kv) * SS * HD,
            Ob + (size_t)b * SS * HH + (size_t)h * HD,
            SS, HD, SS, SS, HD, HH);
    }

    // ---- 6. output projection ----
    cvt(Ob, Obb, (size_t)MTOT * HH);
    gemm_bf16_wmma<<<dim3(HH / BN, MTOT / BM), 128, 0, stream>>>(
        Obb, Wob, out, MTOT, HH, HH, HH, HH, HH);
}
